// WeightedTemporalSelfAttention_63943473103499
// MI455X (gfx1250) — matrix operand order repaired and checked by reading, not compiled, not run on hardware
//
#include <hip/hip_runtime.h>
#include <hip/hip_bf16.h>

// ---------------------------------------------------------------------------
// CDNA5 (gfx1250) wave32 WMMA self-attention. LDS-tiled GEMMs staged by the
// Tensor Data Mover (tensor_load_to_lds, TENSORcnt), double-buffered.
// ---------------------------------------------------------------------------

typedef __attribute__((ext_vector_type(16))) __bf16        v16bf;
typedef __attribute__((ext_vector_type(8)))  float         v8f;
typedef __attribute__((ext_vector_type(4)))  unsigned int  u32x4;
typedef __attribute__((ext_vector_type(4)))  int           s32x4;
typedef __attribute__((ext_vector_type(8)))  int           s32x8;

#define TOKENS 4096   // N tokens = 64*64
#define DKQ    2048
#define DV     2048

#define BM 128        // block C tile rows
#define BN 128        // block C tile cols
#define BK 64         // K slab per LDS stage
#define LDS_STRIDE 72 // 64 + 8 pad: 144B row stride -> 36-bank step, 16B aligned

__device__ __forceinline__ unsigned short f32_to_bf16(float f) {
    unsigned int u = __builtin_bit_cast(unsigned int, f);
    u += 0x7FFFu + ((u >> 16) & 1u);          // round-to-nearest-even
    return (unsigned short)(u >> 16);
}

__device__ __forceinline__ v8f wmma_bf16(v16bf a, v16bf b, v8f c) {
    return __builtin_amdgcn_wmma_f32_16x16x32_bf16(false, a, false, b,
                                                   (short)0, c, false, false);
}

// ---------------------------------------------------------------------------
// TDM: one 2D tile (BK x BM rows) global->LDS DMA.  D# per ISA ch.8:
//  group0: count=1 | lds_addr | global tile addr (57b) | type=2
//  group1: wg_mask=0, data_size=2B, pad_enable, pad_interval=4 (32 DWORDs
//          = one 128B row), pad_amount=3 (4 DWORDs = 16B) -> LDS_STRIDE=72,
//          tensor dims huge (tiles never OOB), tile_dim0=BK, tile_dim1=BM,
//          tensor_dim0_stride = ld (elements)
//  groups 2/3: zero (tile_dim2/3 unused -> 2D)
// ---------------------------------------------------------------------------
__device__ __forceinline__ void tdm_load_tile(const unsigned short* gtile,
                                              unsigned lds_off, unsigned ld_elems) {
    const unsigned long long ga = (unsigned long long)(size_t)gtile;
    s32x4 g0;
    g0[0] = 1;                                               // count=1, user D#
    g0[1] = (int)lds_off;                                    // LDS byte address
    g0[2] = (int)(ga & 0xFFFFFFFFu);                         // global addr lo
    g0[3] = (int)(((ga >> 32) & 0x01FFFFFFu) | (2u << 30));  // addr hi | type=2
    s32x8 g1;
    g1[0] = (1 << 16) | (1 << 20) | (4 << 22) | (3 << 25);   // 2B elems, padding
    g1[1] = 0;                                               // tensor_dim0 lo16<<16
    g1[2] = 0x10;                                            // tensor_dim0=1M, dim1 lo
    g1[3] = 0x10 | (BK << 16);                               // dim1=1M | tile_dim0=64
    g1[4] = BM;                                              // tile_dim1=128, dim2=0
    g1[5] = (int)ld_elems;                                   // dim0 stride lo32
    g1[6] = 0;                                               // stride hi / dim1 stride
    g1[7] = 0;
    const s32x4 gz = {0, 0, 0, 0};                           // groups 2/3: 2D tile
    asm volatile("tensor_load_to_lds %0, %1, %2, %3"
                 :: "s"(g0), "s"(g1), "s"(gz), "s"(gz) : "memory");
}

// ---- WMMA fragment loads from LDS (ISA 7.12.2 layouts, wave32) -------------
// A 16x32: lane<16 rows 0..15 K{0..7,16..23}; lane>=16 same rows K{8..15,24..31}
__device__ __forceinline__ v16bf lds_a_frag(const unsigned short (*As)[LDS_STRIDE],
                                            int mSub, int ks, int lane) {
    const int r = lane & 15, h = lane >> 4;
    const unsigned short* p = &As[mSub + r][ks + h * 8];
    union { u32x4 u[2]; v16bf v; } f;
    f.u[0] = *reinterpret_cast<const u32x4*>(p);
    f.u[1] = *reinterpret_cast<const u32x4*>(p + 16);
    return f.v;
}
// B 32x16 from Bt rows: lane holds column n=lane&15; lane<16 K{0..7,16..23}; lane>=16 K{8..15,24..31}
__device__ __forceinline__ v16bf lds_b_frag(const unsigned short (*Bs)[LDS_STRIDE],
                                            int nSub, int ks, int lane) {
    const int n = lane & 15, h = lane >> 4;
    const unsigned short* p = &Bs[nSub + n][ks + h * 8];
    union { u32x4 u[2]; v16bf v; } f;
    f.u[0] = *reinterpret_cast<const u32x4*>(p);
    f.u[1] = *reinterpret_cast<const u32x4*>(p + 16);
    return f.v;
}

// ---------------------------------------------------------------------------
// Core: C[128x128] = A[M x K] * Bt[N x K]^T  (bf16 in, f32 acc)
// 256 threads = 8 waves; wave (waveM 0..3, waveN 0..1) owns 32x64:
// acc[i*4+t] = C tile (mBlock + waveM*32 + i*16, nBlock + waveN*64 + t*16)
// Wave 0 drives the TDM; TENSORcnt + workgroup barrier synchronize.
// ---------------------------------------------------------------------------
__device__ __forceinline__ void gemm_core(const unsigned short* __restrict__ A, int ldA,
                                          const unsigned short* __restrict__ Bt, int ldB,
                                          int K, int mBlock, int nBlock, v8f acc[8]) {
    __shared__ unsigned short As[2][BM][LDS_STRIDE];
    __shared__ unsigned short Bs[2][BN][LDS_STRIDE];

    const int tid   = threadIdx.x;
    const int lane  = tid & 31;
    const int wave  = tid >> 5;
    const int waveM = wave & 3;
    const int waveN = wave >> 2;
    const int nTile = K / BK;

    const unsigned aOfs0 = (unsigned)(size_t)&As[0][0][0];
    const unsigned aOfs1 = (unsigned)(size_t)&As[1][0][0];
    const unsigned bOfs0 = (unsigned)(size_t)&Bs[0][0][0];
    const unsigned bOfs1 = (unsigned)(size_t)&Bs[1][0][0];

    if (wave == 0) {
        tdm_load_tile(A  + (size_t)mBlock * ldA, aOfs0, (unsigned)ldA);
        tdm_load_tile(Bt + (size_t)nBlock * ldB, bOfs0, (unsigned)ldB);
        __builtin_amdgcn_s_wait_tensorcnt(0);
    }
    __syncthreads();

    for (int kt = 0; kt < nTile; ++kt) {
        const int buf = kt & 1;
        if (wave == 0 && kt + 1 < nTile) {   // DMA next slab into other buffer
            const int kNext = (kt + 1) * BK;
            tdm_load_tile(A  + (size_t)mBlock * ldA + kNext,
                          buf ? aOfs0 : aOfs1, (unsigned)ldA);
            tdm_load_tile(Bt + (size_t)nBlock * ldB + kNext,
                          buf ? bOfs0 : bOfs1, (unsigned)ldB);
        }
#pragma unroll
        for (int ks = 0; ks < BK; ks += 32) {
            v16bf afr[2], bfr[4];
#pragma unroll
            for (int i = 0; i < 2; ++i)
                afr[i] = lds_a_frag(As[buf], waveM * 32 + i * 16, ks, lane);
#pragma unroll
            for (int t = 0; t < 4; ++t)
                bfr[t] = lds_b_frag(Bs[buf], waveN * 64 + t * 16, ks, lane);
#pragma unroll
            for (int i = 0; i < 2; ++i)
#pragma unroll
                for (int t = 0; t < 4; ++t)
                    acc[i * 4 + t] = wmma_bf16(afr[i], bfr[t], acc[i * 4 + t]);
        }
        if (wave == 0)
            __builtin_amdgcn_s_wait_tensorcnt(0);  // next-slab DMA done
        __syncthreads();                           // all done reading buf
    }
}

// ---------------------------------------------------------------------------
// Kernel 1a: x (f32) -> xb (bf16)
// ---------------------------------------------------------------------------
__global__ void cvt_x_kernel(const float* __restrict__ x,
                             unsigned short* __restrict__ xb) {
    size_t i = (size_t)blockIdx.x * blockDim.x + threadIdx.x;
    xb[i] = f32_to_bf16(x[i]);
}

// Kernel 1b: W_{q,k,v} (f32 [K][N]) -> Wt (bf16 [3][N][K]); coalesced reads
__global__ void cvt_w_kernel(const float* __restrict__ Wq,
                             const float* __restrict__ Wk,
                             const float* __restrict__ Wv,
                             unsigned short* __restrict__ Wt) {
    size_t i = (size_t)blockIdx.x * blockDim.x + threadIdx.x;   // 3*2048*2048
    const size_t per = (size_t)DKQ * DKQ;
    const int o = (int)(i / per);
    const size_t rem = i % per;
    const int k = (int)(rem / DKQ);
    const int n = (int)(rem % DKQ);     // consecutive tid -> consecutive n
    const float* W = (o == 0) ? Wq : (o == 1) ? Wk : Wv;
    Wt[(size_t)o * per + (size_t)n * DKQ + k] = f32_to_bf16(W[rem]);
}

// ---------------------------------------------------------------------------
// Kernel 2: fused QKV projection. grid (16, 32, 3), 256 threads.
// q,k row-major bf16; v stored transposed vT[n][m] (packed b128 stores).
// ---------------------------------------------------------------------------
__global__ void gemm_qkv_kernel(const unsigned short* __restrict__ xb,
                                const unsigned short* __restrict__ Wt,
                                unsigned short* __restrict__ qb,
                                unsigned short* __restrict__ kb,
                                unsigned short* __restrict__ vT) {
    const int o = blockIdx.z;
    const int mBlock = blockIdx.y * BM;
    const int nBlock = blockIdx.x * BN;
    v8f acc[8] = {};
    gemm_core(xb, DKQ, Wt + (size_t)o * DKQ * DKQ, DKQ, DKQ, mBlock, nBlock, acc);

    const int lane = threadIdx.x & 31;
    const int wave = threadIdx.x >> 5;
    const int waveM = wave & 3, waveN = wave >> 2;
    const int n0 = lane & 15, h = lane >> 4;
#pragma unroll
    for (int i = 0; i < 2; ++i) {
        const int mSub = mBlock + waveM * 32 + i * 16;
#pragma unroll
        for (int t = 0; t < 4; ++t) {
            const int nSub = nBlock + waveN * 64 + t * 16;
            const v8f a = acc[i * 4 + t];
            if (o < 2) {
                unsigned short* C = (o == 0) ? qb : kb;
#pragma unroll
                for (int j = 0; j < 8; ++j)
                    C[(size_t)(mSub + j + 8 * h) * DKQ + nSub + n0] =
                        f32_to_bf16(a[j]);
            } else {
                union { unsigned short s[8]; u32x4 u; } pk;
#pragma unroll
                for (int j = 0; j < 8; ++j) pk.s[j] = f32_to_bf16(a[j]);
                unsigned short* p = vT + (size_t)(nSub + n0) * TOKENS + mSub + 8 * h;
                *reinterpret_cast<u32x4*>(p) = pk.u;
            }
        }
    }
}

// ---------------------------------------------------------------------------
// Kernel 3: scores = (q @ k^T + same_channel_bias) / sqrt(d_kq). grid (32,32).
// ---------------------------------------------------------------------------
__global__ void gemm_scores_kernel(const unsigned short* __restrict__ qb,
                                   const unsigned short* __restrict__ kb,
                                   float* __restrict__ scores) {
    const int mBlock = blockIdx.y * BM;
    const int nBlock = blockIdx.x * BN;
    v8f acc[8] = {};
    gemm_core(qb, DKQ, kb, DKQ, DKQ, mBlock, nBlock, acc);

    const int lane = threadIdx.x & 31;
    const int wave = threadIdx.x >> 5;
    const int waveM = wave & 3, waveN = wave >> 2;
    const int n0 = lane & 15, h = lane >> 4;
    const float scale = 0.022097086912079608f;   // 1/sqrt(2048)
#pragma unroll
    for (int i = 0; i < 2; ++i)
#pragma unroll
        for (int t = 0; t < 4; ++t)
#pragma unroll
            for (int j = 0; j < 8; ++j) {
                const int m = mBlock + waveM * 32 + i * 16 + j + 8 * h;
                const int n = nBlock + waveN * 64 + t * 16 + n0;
                const float bias = ((m >> 6) == (n >> 6)) ? 1.0f : 0.0f;
                scores[(size_t)m * TOKENS + n] = (acc[i * 4 + t][j] + bias) * scale;
            }
}

// ---------------------------------------------------------------------------
// Kernel 4: row softmax, f32 scores -> bf16 attn. One block (256 thr) per row.
// ---------------------------------------------------------------------------
__global__ void softmax_kernel(const float* __restrict__ scores,
                               unsigned short* __restrict__ attn) {
    __shared__ float red[256];
    const int row = blockIdx.x;
    const int tid = threadIdx.x;
    const float* s = scores + (size_t)row * TOKENS;

    float m = -3.402823466e38f;
    for (int i = tid; i < TOKENS; i += 256) m = fmaxf(m, s[i]);
    red[tid] = m;
    __syncthreads();
    for (int off = 128; off > 0; off >>= 1) {
        if (tid < off) red[tid] = fmaxf(red[tid], red[tid + off]);
        __syncthreads();
    }
    const float rowmax = red[0];
    __syncthreads();

    float sum = 0.0f;
    for (int i = tid; i < TOKENS; i += 256) sum += __expf(s[i] - rowmax);
    red[tid] = sum;
    __syncthreads();
    for (int off = 128; off > 0; off >>= 1) {
        if (tid < off) red[tid] += red[tid + off];
        __syncthreads();
    }
    const float inv = 1.0f / red[0];

    for (int i = tid; i < TOKENS; i += 256)
        attn[(size_t)row * TOKENS + i] = f32_to_bf16(__expf(s[i] - rowmax) * inv);
}

// ---------------------------------------------------------------------------
// Kernel 5: ctx = attn @ v + b_channel + b_t[0] + pe_row(0). grid (16, 32).
// ---------------------------------------------------------------------------
__global__ void gemm_ctx_kernel(const unsigned short* __restrict__ attn,
                                const unsigned short* __restrict__ vT,
                                const float* __restrict__ b_channel,
                                const float* __restrict__ b_t,
                                float* __restrict__ out) {
    const int mBlock = blockIdx.y * BM;
    const int nBlock = blockIdx.x * BN;
    v8f acc[8] = {};
    gemm_core(attn, TOKENS, vT, TOKENS, TOKENS, mBlock, nBlock, acc);

    const int lane = threadIdx.x & 31;
    const int wave = threadIdx.x >> 5;
    const int waveM = wave & 3, waveN = wave >> 2;
    const int n0 = lane & 15, h = lane >> 4;
#pragma unroll
    for (int i = 0; i < 2; ++i)
#pragma unroll
        for (int t = 0; t < 4; ++t)
#pragma unroll
            for (int j = 0; j < 8; ++j) {
                const int m = mBlock + waveM * 32 + i * 16 + j + 8 * h;
                const int n = nBlock + waveN * 64 + t * 16 + n0;
                const size_t idx = (size_t)m * DV + n;
                // pe_row(t=0): ang = 0 -> sin=0 (even n), cos=1 (odd n)
                const float pe = (n & 1) ? 1.0f : 0.0f;
                out[idx] = acc[i * 4 + t][j] + b_channel[idx] + b_t[idx] + pe;
            }
}

// ---------------------------------------------------------------------------
// Launcher
// ---------------------------------------------------------------------------
extern "C" void kernel_launch(void* const* d_in, const int* in_sizes, int n_in,
                              void* d_out, int out_size, void* d_ws, size_t ws_size,
                              hipStream_t stream) {
    const float* x   = (const float*)d_in[0];
    const float* Wq  = (const float*)d_in[1];
    const float* Wk  = (const float*)d_in[2];
    const float* Wv  = (const float*)d_in[3];
    const float* b_c = (const float*)d_in[4];
    const float* b_t = (const float*)d_in[5];   // step 0 slice
    float* out = (float*)d_out;

    unsigned short* xb  = (unsigned short*)d_ws;                 // 4096*2048
    unsigned short* Wt  = xb  + (size_t)TOKENS * DKQ;            // 3*2048*2048
    unsigned short* qb  = Wt  + (size_t)3 * DKQ * DKQ;           // 4096*2048
    unsigned short* kb  = qb  + (size_t)TOKENS * DKQ;            // 4096*2048
    unsigned short* vT  = kb  + (size_t)TOKENS * DKQ;            // 2048*4096
    float*          sc  = (float*)(vT + (size_t)DV * TOKENS);    // 4096*4096 f32
    unsigned short* att = (unsigned short*)(sc + (size_t)TOKENS * TOKENS);

    cvt_x_kernel<<<(TOKENS * DKQ) / 256, 256, 0, stream>>>(x, xb);
    cvt_w_kernel<<<(3 * DKQ * DKQ) / 256, 256, 0, stream>>>(Wq, Wk, Wv, Wt);
    gemm_qkv_kernel<<<dim3(DKQ / BN, TOKENS / BM, 3), 256, 0, stream>>>(xb, Wt, qb, kb, vT);
    gemm_scores_kernel<<<dim3(TOKENS / BN, TOKENS / BM), 256, 0, stream>>>(qb, kb, sc);
    softmax_kernel<<<TOKENS, 256, 0, stream>>>(sc, att);
    gemm_ctx_kernel<<<dim3(DV / BN, TOKENS / BM), 256, 0, stream>>>(att, vT, b_c, b_t, out);
}
